// ResKmeans_6949257085596
// MI455X (gfx1250) — matrix operand order repaired and checked by reading, not compiled, not run on hardware
//
#include <hip/hip_runtime.h>

#define N_PTS   32768
#define L_LAYERS 8
#define K_CENT  8192
#define D_DIM   512

typedef __attribute__((ext_vector_type(16))) __bf16 v16bf;
typedef __attribute__((ext_vector_type(8)))  __bf16 v8bf;
typedef __attribute__((ext_vector_type(8)))  float  v8f;

__device__ __forceinline__ v8f wmma_bf16(v16bf a, v16bf b, v8f c) {
  // 8 args: (neg_a, A, neg_b, B, c_mod, C, reuse_a, reuse_b)
  return __builtin_amdgcn_wmma_f32_16x16x32_bf16(false, a, false, b, (short)0, c, false, false);
}

__device__ __forceinline__ v16bf join8(v8bf a, v8bf b) {
  return __builtin_shufflevector(a, b, 0,1,2,3,4,5,6,7,8,9,10,11,12,13,14,15);
}

// ---------------------------------------------------------------- init: x -> R
__global__ __launch_bounds__(256) void rk_copy_x(const float4* __restrict__ x,
                                                 float4* __restrict__ R, int n4) {
  for (int i = blockIdx.x * blockDim.x + threadIdx.x; i < n4; i += gridDim.x * blockDim.x)
    R[i] = x[i];
}

// ------------------------------------------- per-layer: split centroids + norms
// Writes Bhi/Blo as B fragments in this kernel's contiguous k order (matches A): for centroid chunk g (16 centroids),
// d-chunk c (32 d): lanes 0-15 hold K=0-15, lanes 16-31 hold K=16-31,
// packed element index = k mod 16.
__global__ __launch_bounds__(256) void rk_prep(const float* __restrict__ cent,
                                               unsigned short* __restrict__ bhi_,
                                               unsigned short* __restrict__ blo_,
                                               float* __restrict__ cnorm) {
  __bf16* Bhi = (__bf16*)bhi_;
  __bf16* Blo = (__bf16*)blo_;
  __shared__ float red[256];
  const int k = blockIdx.x, tid = threadIdx.x;
  const int g = k >> 4, n = k & 15;
  const float* cp = cent + (size_t)k * D_DIM;
  float ss = 0.f;
  for (int d = tid; d < D_DIM; d += 256) {
    float v = cp[d];
    ss += v * v;
    __bf16 h = (__bf16)v;
    __bf16 l = (__bf16)(v - (float)h);
    const int c = d >> 5, kk = d & 31;
    const int lane = n + ((kk >> 4) << 4);
    const size_t idx = ((size_t)((g * 16 + c) * 32 + lane)) * 16 + (kk & 15);
    Bhi[idx] = h;
    Blo[idx] = l;
  }
  red[tid] = ss;
  __syncthreads();
  for (int s = 128; s > 0; s >>= 1) {
    if (tid < s) red[tid] += red[tid + s];
    __syncthreads();
  }
  if (tid == 0) cnorm[k] = red[0];
}

// -------------------------- per-layer fused: distances + argmin + residual update
// Grid: N/32 workgroups of 256 threads (8 waves). Each wave owns a distinct
// 32-centroid block per iteration (2 fragment groups) and two 16-row tiles:
// 2x2 register blocking -> 4 independent WMMA accumulation chains.
__global__ __launch_bounds__(256) void rk_layer(
    float* __restrict__ R,
    const unsigned short* __restrict__ bhi_,
    const unsigned short* __restrict__ blo_,
    const float* __restrict__ cnorm,
    const float* __restrict__ cent,
    int* __restrict__ codes, int layer) {
  __shared__ __align__(16) unsigned char smem[65536];  // Ahi|Alo, later reused for reduce
  __bf16* Ahi = (__bf16*)smem;
  __bf16* Alo = Ahi + 32 * D_DIM;

  const int tid  = threadIdx.x;
  const int wid  = tid >> 5, lane = tid & 31;
  const int half = lane >> 4, m = lane & 15;
  const int rowBase = blockIdx.x * 32;

  // Stage 32 residual rows into LDS as bf16 hi/lo (row-major by d)
  {
    const float4* Rv = (const float4*)(R + (size_t)rowBase * D_DIM);
    for (int j = tid; j < (32 * D_DIM) / 4; j += 256) {
      float4 v = Rv[j];
      float vs[4] = {v.x, v.y, v.z, v.w};
      const int e = j * 4;
#pragma unroll
      for (int u = 0; u < 4; ++u) {
        __bf16 h = (__bf16)vs[u];
        Ahi[e + u] = h;
        Alo[e + u] = (__bf16)(vs[u] - (float)h);
      }
    }
  }
  __syncthreads();

  float bestV[2][8];
  int   bestI[2][8];
#pragma unroll
  for (int rb = 0; rb < 2; ++rb)
#pragma unroll
    for (int r = 0; r < 8; ++r) { bestV[rb][r] = 3.4e38f; bestI[rb][r] = 0; }

  const v16bf* BH = (const v16bf*)bhi_;
  const v16bf* BL = (const v16bf*)blo_;

  for (int it = 0; it < K_CENT / 256; ++it) {      // 32 iterations
    const int k0 = it * 256 + wid * 32;            // this wave's 32 centroids
    const int g  = k0 >> 4;                        // fragment groups g, g+1

    // Prefetch next iteration's B fragment lines (L2-resident stream -> L0)
    if (it + 1 < K_CENT / 256) {
      const int gn = (k0 + 256) >> 4;
      __builtin_prefetch(&BH[(size_t)(gn * 16) * 32 + lane], 0, 0);
      __builtin_prefetch(&BL[(size_t)(gn * 16) * 32 + lane], 0, 0);
    }

    v8f acc00 = {};  // [rowBlk0][centBlk0]
    v8f acc01 = {};  // [rowBlk0][centBlk1]
    v8f acc10 = {};  // [rowBlk1][centBlk0]
    v8f acc11 = {};  // [rowBlk1][centBlk1]
#pragma unroll 4
    for (int c = 0; c < 16; ++c) {                 // D chunks of 32
      const int f0 = (g * 16 + c) * 32 + lane;
      const int f1 = f0 + 16 * 32;                 // group g+1
      v16bf bh0 = BH[f0];                          // 32B = 2x global_load_b128
      v16bf bl0 = BL[f0];
      v16bf bh1 = BH[f1];
      v16bf bl1 = BL[f1];

      // A layout (matches B packing): lanes 0-15 K=0-15; lanes 16-31 K=16-31
      const int base0 = m * D_DIM + c * 32 + half * 16;         // rows 0..15
      const int base1 = (16 + m) * D_DIM + c * 32 + half * 16;  // rows 16..31
      v16bf ah0 = join8(*(const v8bf*)(Ahi + base0), *(const v8bf*)(Ahi + base0 + 8));
      v16bf al0 = join8(*(const v8bf*)(Alo + base0), *(const v8bf*)(Alo + base0 + 8));
      v16bf ah1 = join8(*(const v8bf*)(Ahi + base1), *(const v8bf*)(Ahi + base1 + 8));
      v16bf al1 = join8(*(const v8bf*)(Alo + base1), *(const v8bf*)(Alo + base1 + 8));

      // bf16 hi/lo split GEMM: dot ~= hi*hi + hi*lo + lo*hi (4 independent chains)
      acc00 = wmma_bf16(ah0, bh0, acc00);
      acc01 = wmma_bf16(ah0, bh1, acc01);
      acc10 = wmma_bf16(ah1, bh0, acc10);
      acc11 = wmma_bf16(ah1, bh1, acc11);
      acc00 = wmma_bf16(ah0, bl0, acc00);
      acc01 = wmma_bf16(ah0, bl1, acc01);
      acc10 = wmma_bf16(ah1, bl0, acc10);
      acc11 = wmma_bf16(ah1, bl1, acc11);
      acc00 = wmma_bf16(al0, bh0, acc00);
      acc01 = wmma_bf16(al0, bh1, acc01);
      acc10 = wmma_bf16(al1, bh0, acc10);
      acc11 = wmma_bf16(al1, bh1, acc11);
    }
    // score = ||c||^2 - 2*dot  (||r||^2 constant per row, irrelevant for argmin)
    const float cn0 = cnorm[k0 + m];
    const float cn1 = cnorm[k0 + 16 + m];
    const int   kg0 = k0 + m;
    const int   kg1 = k0 + 16 + m;
#pragma unroll
    for (int r = 0; r < 8; ++r) {
      float s;
      s = __builtin_fmaf(-2.f, acc00[r], cn0);
      if (s < bestV[0][r]) { bestV[0][r] = s; bestI[0][r] = kg0; }
      s = __builtin_fmaf(-2.f, acc01[r], cn1);
      if (s < bestV[0][r]) { bestV[0][r] = s; bestI[0][r] = kg1; }
      s = __builtin_fmaf(-2.f, acc10[r], cn0);
      if (s < bestV[1][r]) { bestV[1][r] = s; bestI[1][r] = kg0; }
      s = __builtin_fmaf(-2.f, acc11[r], cn1);
      if (s < bestV[1][r]) { bestV[1][r] = s; bestI[1][r] = kg1; }
    }
  }

  __syncthreads();  // done reading A tiles; reuse smem for reductions
  float* sVal  = (float*)smem;          // [8 waves][32 rows]
  int*   sIdx  = (int*)(smem + 1024);   // [8][32]
  int*   sCode = (int*)(smem + 2048);   // [32]

  // C/D layout: VGPR r, lanes 0-15 -> M=r; lanes 16-31 -> M=8+r; N = lane&15.
  // Reduce over N (16 lanes within each half), prefer lower index on ties.
#pragma unroll
  for (int rb = 0; rb < 2; ++rb)
#pragma unroll
    for (int r = 0; r < 8; ++r) {
      float v = bestV[rb][r];
      int   i = bestI[rb][r];
#pragma unroll
      for (int off = 8; off >= 1; off >>= 1) {
        float v2 = __shfl_xor(v, off, 16);
        int   i2 = __shfl_xor(i, off, 16);
        if (v2 < v || (v2 == v && i2 < i)) { v = v2; i = i2; }
      }
      if (m == 0) {
        const int row = rb * 16 + half * 8 + r;
        sVal[wid * 32 + row] = v;
        sIdx[wid * 32 + row] = i;
      }
    }
  __syncthreads();

  if (tid < 32) {
    float v = sVal[tid];
    int   i = sIdx[tid];
    for (int w = 1; w < 8; ++w) {
      float v2 = sVal[w * 32 + tid];
      int   i2 = sIdx[w * 32 + tid];
      if (v2 < v || (v2 == v && i2 < i)) { v = v2; i = i2; }
    }
    sCode[tid] = i;
    codes[(size_t)(rowBase + tid) * L_LAYERS + layer] = i;
  }
  __syncthreads();

  // Residual update in exact fp32: r -= c[code]
  for (int row = wid; row < 32; row += 8) {
    const int code = sCode[row];
    const float* cp = cent + (size_t)code * D_DIM;
    float* rp = R + (size_t)(rowBase + row) * D_DIM;
    for (int d = lane; d < D_DIM; d += 32) rp[d] -= cp[d];
  }
}

// ---------------------------------------------------------------- launcher
extern "C" void kernel_launch(void* const* d_in, const int* in_sizes, int n_in,
                              void* d_out, int out_size, void* d_ws, size_t ws_size,
                              hipStream_t stream) {
  (void)in_sizes; (void)n_in; (void)out_size; (void)ws_size;
  const float* x     = (const float*)d_in[0];                 // [N, D] f32
  const float* cents = (const float*)d_in[1];                 // [L, K, D] f32
  int* codes = (int*)d_out;                                   // [N, L] int32

  char* ws = (char*)d_ws;
  // ws layout: R 64MB | Bhi 8MB | Blo 8MB | cnorm 32KB   (needs ~80.1 MB)
  float*          R     = (float*)ws;
  unsigned short* Bhi   = (unsigned short*)(ws + ((size_t)64 << 20));
  unsigned short* Blo   = (unsigned short*)(ws + ((size_t)72 << 20));
  float*          cnorm = (float*)(ws + ((size_t)80 << 20));

  rk_copy_x<<<2048, 256, 0, stream>>>((const float4*)x, (float4*)R,
                                      (N_PTS * D_DIM) / 4);
  for (int l = 0; l < L_LAYERS; ++l) {
    const float* cl = cents + (size_t)l * K_CENT * D_DIM;
    rk_prep<<<K_CENT, 256, 0, stream>>>(cl, Bhi, Blo, cnorm);
    rk_layer<<<N_PTS / 32, 256, 0, stream>>>(R, Bhi, Blo, cnorm, cl, codes, l);
  }
}
